// MoE_74105365725750
// MI455X (gfx1250) — matrix operand order repaired and checked by reading, not compiled, not run on hardware
//
#include <hip/hip_runtime.h>

// ---------------------------------------------------------------------------
// CDNA5 WMMA types (wave32, v_wmma_f32_16x16x32_bf16)
// ---------------------------------------------------------------------------
typedef __attribute__((ext_vector_type(16))) __bf16 v16bf;
typedef __attribute__((ext_vector_type(8)))  float  v8f;
typedef int v4i_vs __attribute__((vector_size(16)));   // matches async-LDS builtin pointee

__device__ __forceinline__ unsigned short f2bf(float f) {
    unsigned u = __float_as_uint(f);
    u += 0x7FFFu + ((u >> 16) & 1u);     // round-to-nearest-even
    return (unsigned short)(u >> 16);
}

// Guarded CDNA5 async global->LDS copy (falls back to plain b128 copy).
#if defined(__has_builtin)
#  if __has_builtin(__builtin_amdgcn_global_load_async_to_lds_b128) && \
      __has_builtin(__builtin_amdgcn_s_wait_asynccnt)
#    define USE_ASYNC_LDS 1
#  endif
#endif
#ifndef USE_ASYNC_LDS
#  define USE_ASYNC_LDS 0
#endif

__device__ __forceinline__ void copy16B_g2l(const unsigned short* __restrict__ gsrc,
                                            unsigned short* ldst) {
#if USE_ASYNC_LDS
    __builtin_amdgcn_global_load_async_to_lds_b128(
        (__attribute__((address_space(1))) v4i_vs*)gsrc,
        (__attribute__((address_space(3))) v4i_vs*)ldst, 0, 0);
#else
    *(uint4*)ldst = *(const uint4*)gsrc;
#endif
}

__device__ __forceinline__ void wait_async_then_sync() {
#if USE_ASYNC_LDS
    __builtin_amdgcn_s_wait_asynccnt(0);
#endif
    __syncthreads();
}

// ---------------------------------------------------------------------------
// Weight image builder: writes B operand as bf16 in fragment order
//   img[((nb*KSE + ks)*4 + frag)*512 + lane*16 + j]
// B fragment map (32x16): col=lane&15, K=(j>>3)*16 + (lane>>4)*8 + (j&7)
// bt=1: B stored [N][K] (conv OIHW flat, cls_w); bt=0: B stored [K][N] (g1w,g2w)
// ---------------------------------------------------------------------------
__global__ void build_bimg(const float* __restrict__ B, unsigned short* __restrict__ img,
                           int N, int K, int KSE, int bt)
{
    int idx  = blockIdx.x * 256 + threadIdx.x;
    int j    = idx & 15;
    int ln   = (idx >> 4) & 31;
    int frag = (idx >> 9) & 3;
    int blk  = idx >> 11;
    int ks   = blk % KSE;
    int nb   = blk / KSE;
    int n = nb * 64 + frag * 16 + (ln & 15);
    int k = ks * 32 + ((j >> 3) << 4) + ((ln >> 4) << 3) + (j & 7);
    float v = 0.f;
    if (n < N && k < K) v = bt ? B[n * K + k] : B[k * N + n];
    img[idx] = f2bf(v);
}

// ---------------------------------------------------------------------------
// Fused Conv3x3(SAME)+bias+BN(eval)+ReLU as implicit GEMM.
// C[M,N] = A[M,K] x B[K,N];  M = B*H*W, N = Cout, K = Cin*9.
// Block tile 128(M) x 64(N); 8 waves as 4(m) x 2(n); each wave: 2x2 fragments.
// K unrolled by 2 (64 K per staging barrier) -> 8 WMMA per barrier per wave.
// Weights come from pre-built bf16 fragment image (pure async copy to LDS).
// ---------------------------------------------------------------------------
template<int H, int W, int Cin, int Cout>
__global__ __launch_bounds__(256)
void conv_bn_relu_wmma(const float* __restrict__ x,            // (B,Cin,H,W) f32
                       const unsigned short* __restrict__ wimg,// bf16 fragment image
                       const float* __restrict__ cb,
                       const float* __restrict__ bg, const float* __restrict__ bb,
                       const float* __restrict__ bm, const float* __restrict__ bv,
                       float* __restrict__ out)                // (B,Cout,H,W) f32
{
    constexpr int K   = Cin * 9;
    constexpr int KSE = (((K + 31) / 32) + 1) & ~1;   // even # of 32-K steps

    const int m0 = blockIdx.x * 128;
    const int nb = blockIdx.y;
    const int n0 = nb * 64;

    __shared__ __align__(32) unsigned short Asw[2 * 8 * 32 * 16]; // [u][mfrag][lane][j] 16KB
    __shared__ __align__(32) unsigned short Bsw[2 * 4 * 32 * 16]; // [u][nfrag][lane][j]  8KB

    const int tid  = threadIdx.x;
    const int lane = tid & 31;
    const int wave = tid >> 5;
    const int wm   = wave >> 1;   // 0..3
    const int wn   = wave & 1;    // 0..1

    v8f acc[2][2];
#pragma unroll
    for (int a = 0; a < 2; ++a)
#pragma unroll
        for (int b = 0; b < 2; ++b)
            acc[a][b] = v8f{0.f,0.f,0.f,0.f,0.f,0.f,0.f,0.f};

    // A-staging: this thread owns one k column (of 64) and 32 of 128 rows.
    const int kk64 = tid & 63;
    const int rB   = (tid >> 6) * 32;
    const int kst  = kk64 >> 5;
    const int kk   = kk64 & 31;
    const int half = (kk >> 3) & 1;                    // A frag k map (ISA 7.12.2)
    const int jj   = (kk & 7) + ((kk >> 4) << 3);

    for (int s2 = 0; s2 < KSE / 2; ++s2) {
        __syncthreads();
        // ---- B: straight copy of 4096 bf16 from fragment image (async) ----
        {
            const unsigned short* src = wimg + (size_t)(nb * KSE + s2 * 2) * 2048 + tid * 16;
            unsigned short* dst = Bsw + tid * 16;
            copy16B_g2l(src, dst);
            copy16B_g2l(src + 8, dst + 8);
            if (s2 + 1 < KSE / 2)
                __builtin_prefetch(src + 4096, 0, 1);   // global_prefetch_b8
        }
        // ---- A: im2col 128x64, k-decode hoisted per thread ----
        {
            const int k = s2 * 64 + kk64;
            const bool valid = (k < K);
            int ci = 0, rr = 0, sc = 0;
            if (valid) { ci = k / 9; int t = k - ci * 9; rr = t / 3; sc = t - rr * 3; }
            for (int r2 = 0; r2 < 32; ++r2) {
                const int row = rB + r2;
                const int m   = m0 + row;
                float v = 0.f;
                if (valid) {
                    int bi = m / (H * W);            // power-of-two -> shifts
                    int oh = (m / W) % H;
                    int ow = m % W;
                    int ih = oh + rr - 1, iw = ow + sc - 1;
                    if (ih >= 0 && ih < H && iw >= 0 && iw < W)
                        v = x[((bi * Cin + ci) * H + ih) * W + iw];
                }
                Asw[(((kst * 8 + (row >> 4)) * 32) + half * 16 + (row & 15)) * 16 + jj] = f2bf(v);
            }
        }
        wait_async_then_sync();

#pragma unroll
        for (int u = 0; u < 2; ++u) {
            v16bf a0 = *reinterpret_cast<const v16bf*>(&Asw[((u * 8 + wm * 2 + 0) * 32 + lane) * 16]);
            v16bf a1 = *reinterpret_cast<const v16bf*>(&Asw[((u * 8 + wm * 2 + 1) * 32 + lane) * 16]);
            v16bf b0 = *reinterpret_cast<const v16bf*>(&Bsw[((u * 4 + wn * 2 + 0) * 32 + lane) * 16]);
            v16bf b1 = *reinterpret_cast<const v16bf*>(&Bsw[((u * 4 + wn * 2 + 1) * 32 + lane) * 16]);
            acc[0][0] = __builtin_amdgcn_wmma_f32_16x16x32_bf16(false, a0, false, b0, (short)0, acc[0][0], false, false);
            acc[0][1] = __builtin_amdgcn_wmma_f32_16x16x32_bf16(false, a0, false, b1, (short)0, acc[0][1], false, false);
            acc[1][0] = __builtin_amdgcn_wmma_f32_16x16x32_bf16(false, a1, false, b0, (short)0, acc[1][0], false, false);
            acc[1][1] = __builtin_amdgcn_wmma_f32_16x16x32_bf16(false, a1, false, b1, (short)0, acc[1][1], false, false);
        }
    }

    // ---- epilogue: fold BN + conv bias + ReLU, scatter NCHW ----
    const int col = lane & 15;
#pragma unroll
    for (int ni = 0; ni < 2; ++ni) {
        const int ch = n0 + (wn * 2 + ni) * 16 + col;
        if (ch < Cout) {
            const float inv  = bg[ch] * rsqrtf(bv[ch] + 1e-5f);
            const float bias = cb[ch] * inv + bb[ch] - bm[ch] * inv;
#pragma unroll
            for (int mi = 0; mi < 2; ++mi) {
                const int mBase = m0 + (wm * 2 + mi) * 16 + ((lane >> 4) << 3);
#pragma unroll
                for (int i = 0; i < 8; ++i) {
                    int m  = mBase + i;
                    int bi = m / (H * W);
                    int oh = (m / W) % H;
                    int ow = m % W;
                    float v = acc[mi][ni][i] * inv + bias;
                    v = v > 0.f ? v : 0.f;
                    out[((bi * Cout + ch) * H + oh) * W + ow] = v;
                }
            }
        }
    }
}

// ---------------------------------------------------------------------------
// Generic WMMA GEMM with pre-built B fragment image: C = [relu](A*B + bias)
// ---------------------------------------------------------------------------
template<bool RELU>
__global__ __launch_bounds__(256)
void gemm_wmma(const float* __restrict__ A, const unsigned short* __restrict__ wimg,
               const float* __restrict__ bias, float* __restrict__ C,
               int M, int N, int K, int KSE)
{
    const int m0 = blockIdx.x * 128;
    const int nb = blockIdx.y;
    const int n0 = nb * 64;

    __shared__ __align__(32) unsigned short Asw[2 * 8 * 32 * 16];
    __shared__ __align__(32) unsigned short Bsw[2 * 4 * 32 * 16];

    const int tid  = threadIdx.x;
    const int lane = tid & 31;
    const int wave = tid >> 5;
    const int wm   = wave >> 1;
    const int wn   = wave & 1;

    v8f acc[2][2];
#pragma unroll
    for (int a = 0; a < 2; ++a)
#pragma unroll
        for (int b = 0; b < 2; ++b)
            acc[a][b] = v8f{0.f,0.f,0.f,0.f,0.f,0.f,0.f,0.f};

    const int kk64 = tid & 63;
    const int rB   = (tid >> 6) * 32;
    const int kst  = kk64 >> 5;
    const int kk   = kk64 & 31;
    const int half = (kk >> 3) & 1;
    const int jj   = (kk & 7) + ((kk >> 4) << 3);

    for (int s2 = 0; s2 < KSE / 2; ++s2) {
        __syncthreads();
        {
            const unsigned short* src = wimg + (size_t)(nb * KSE + s2 * 2) * 2048 + tid * 16;
            unsigned short* dst = Bsw + tid * 16;
            copy16B_g2l(src, dst);
            copy16B_g2l(src + 8, dst + 8);
            if (s2 + 1 < KSE / 2)
                __builtin_prefetch(src + 4096, 0, 1);
        }
        {
            const int k = s2 * 64 + kk64;
            const bool kv = (k < K);
            for (int r2 = 0; r2 < 32; ++r2) {
                const int row = rB + r2;
                const int m   = m0 + row;
                float v = (kv && m < M) ? A[m * K + k] : 0.f;
                Asw[(((kst * 8 + (row >> 4)) * 32) + half * 16 + (row & 15)) * 16 + jj] = f2bf(v);
            }
        }
        wait_async_then_sync();

#pragma unroll
        for (int u = 0; u < 2; ++u) {
            v16bf a0 = *reinterpret_cast<const v16bf*>(&Asw[((u * 8 + wm * 2 + 0) * 32 + lane) * 16]);
            v16bf a1 = *reinterpret_cast<const v16bf*>(&Asw[((u * 8 + wm * 2 + 1) * 32 + lane) * 16]);
            v16bf b0 = *reinterpret_cast<const v16bf*>(&Bsw[((u * 4 + wn * 2 + 0) * 32 + lane) * 16]);
            v16bf b1 = *reinterpret_cast<const v16bf*>(&Bsw[((u * 4 + wn * 2 + 1) * 32 + lane) * 16]);
            acc[0][0] = __builtin_amdgcn_wmma_f32_16x16x32_bf16(false, a0, false, b0, (short)0, acc[0][0], false, false);
            acc[0][1] = __builtin_amdgcn_wmma_f32_16x16x32_bf16(false, a0, false, b1, (short)0, acc[0][1], false, false);
            acc[1][0] = __builtin_amdgcn_wmma_f32_16x16x32_bf16(false, a1, false, b0, (short)0, acc[1][0], false, false);
            acc[1][1] = __builtin_amdgcn_wmma_f32_16x16x32_bf16(false, a1, false, b1, (short)0, acc[1][1], false, false);
        }
    }

    const int col = lane & 15;
#pragma unroll
    for (int ni = 0; ni < 2; ++ni) {
        const int ch = n0 + (wn * 2 + ni) * 16 + col;
        if (ch < N) {
            const float bsv = bias ? bias[ch] : 0.f;
#pragma unroll
            for (int mi = 0; mi < 2; ++mi) {
                const int mBase = m0 + (wm * 2 + mi) * 16 + ((lane >> 4) << 3);
#pragma unroll
                for (int i = 0; i < 8; ++i) {
                    int m = mBase + i;
                    if (m < M) {
                        float v = acc[mi][ni][i] + bsv;
                        if (RELU) v = v > 0.f ? v : 0.f;
                        C[m * N + ch] = v;
                    }
                }
            }
        }
    }
}

// ---------------------------------------------------------------------------
// 2x2 max pool (NCHW), stride 2
// ---------------------------------------------------------------------------
__global__ void maxpool2(const float* __restrict__ in, float* __restrict__ out,
                         int C, int H, int W, int total)
{
    int idx = blockIdx.x * blockDim.x + threadIdx.x;
    if (idx >= total) return;
    int HW2 = (H / 2) * (W / 2);
    int ow = idx % (W / 2);
    int oh = (idx / (W / 2)) % (H / 2);
    int c  = (idx / HW2) % C;
    int b  = idx / (HW2 * C);
    const float* p = in + ((b * C + c) * H + 2 * oh) * W + 2 * ow;
    out[idx] = fmaxf(fmaxf(p[0], p[1]), fmaxf(p[W], p[W + 1]));
}

// ---------------------------------------------------------------------------
// Global average pool over 8x8 -> feats (B,256)
// ---------------------------------------------------------------------------
__global__ void avgpool8(const float* __restrict__ in, float* __restrict__ feats)
{
    int idx = blockIdx.x * blockDim.x + threadIdx.x;
    const float* p = in + (size_t)idx * 64;
    float s = 0.f;
#pragma unroll
    for (int i = 0; i < 64; ++i) s += p[i];
    feats[idx] = s * (1.f / 64.f);
}

// ---------------------------------------------------------------------------
// Routing: one workgroup of 1024 threads; (1024x16) score matrix in LDS.
// ---------------------------------------------------------------------------
__global__ __launch_bounds__(1024)
void routing_kernel(const float* __restrict__ gate_logits, // (1024,16)
                    const float* __restrict__ logits_e,    // (1024,160)
                    float* __restrict__ out)               // final|ragg|D
{
    extern __shared__ float smem[];
    float* s_base  = smem;                   // 1024*16
    float* s_pen   = smem + 16384;           // 16
    float* s_load  = smem + 16400;           // 16
    int*   s_ncl   = (int*)(smem + 16416);   // 16
    int*   s_avail = (int*)(smem + 16432);   // 1024
    int*   s_curb  = (int*)(smem + 17456);   // 1

    const int b = threadIdx.x;
    const float* gl = gate_logits + b * 16;
    const float* le = logits_e + b * 160;

    float init[16];
    {
        float mx = -3.4e38f;
#pragma unroll
        for (int e = 0; e < 16; ++e) { init[e] = gl[e]; mx = fmaxf(mx, init[e]); }
        float s = 0.f;
#pragma unroll
        for (int e = 0; e < 16; ++e) { init[e] = __expf(init[e] - mx); s += init[e]; }
        float inv = 1.f / s;
#pragma unroll
        for (int e = 0; e < 16; ++e) init[e] *= inv;
    }

    float base_b[16];
#pragma unroll
    for (int e = 0; e < 16; ++e) {
        float l[10];
        float mx = -3.4e38f;
#pragma unroll
        for (int c = 0; c < 10; ++c) { l[c] = le[e * 10 + c]; mx = fmaxf(mx, l[c]); }
        float s = 0.f;
#pragma unroll
        for (int c = 0; c < 10; ++c) { l[c] = __expf(l[c] - mx); s += l[c]; }
        float inv = 1.f / s;
        float ent = 0.f;
#pragma unroll
        for (int c = 0; c < 10; ++c) {
            float p = l[c] * inv;
            ent -= p * __logf(fmaxf(p, 1e-12f));
        }
        float conf = -ent * 0.43429448190325176f;
        base_b[e] = init[e] * conf;
        s_base[b * 16 + e] = base_b[e];
    }

    bool  avail_b = true;
    int   Dmask   = 0;
    float ragg_b[16];
#pragma unroll
    for (int e = 0; e < 16; ++e) ragg_b[e] = 0.f;

    s_avail[b] = 1;
    if (b < 16) s_load[b] = 0.f;
    __syncthreads();

    for (int iter = 0; iter < 3; ++iter) {
        if (b == 0) s_curb[0] = 0;
        __syncthreads();
        atomicAdd(s_curb, avail_b ? 1 : 0);
        __syncthreads();
        if (b < 16) {
            float ld = s_load[b];
            float fullness = fminf(fmaxf(ld / 128.0f, 0.f), 1.f);
            s_pen[b] = fmaxf(1.f - fullness * 0.1f, 0.f);
            int nc = (int)floorf((128.0f - ld) * 0.5f);
            nc = min(max(nc, 0), s_curb[0]);
            s_ncl[b] = nc;
        }
        __syncthreads();

        float sc[16];
#pragma unroll
        for (int e = 0; e < 16; ++e)
            sc[e] = avail_b ? base_b[e] * s_pen[e] : -3.4e38f;
        int rank[16];
#pragma unroll
        for (int e = 0; e < 16; ++e) rank[e] = 0;
        for (int bp = 0; bp < 1024; ++bp) {
            bool ap = s_avail[bp] != 0;
#pragma unroll
            for (int e = 0; e < 16; ++e) {
                float sp = ap ? s_base[bp * 16 + e] * s_pen[e] : -3.4e38f;
                if (sp > sc[e] || (sp == sc[e] && bp < b)) rank[e]++;
            }
        }

        float mxv = 0.f; int best = 0;
#pragma unroll
        for (int e = 0; e < 16; ++e) {
            bool claimed = avail_b && (rank[e] < s_ncl[e]);
            float cf = claimed ? sc[e] : 0.f;
            if (e == 0) { mxv = cf; best = 0; }
            else if (cf > mxv) { mxv = cf; best = e; }
        }
        if (mxv > 0.f) {
            Dmask |= (1 << best);
            ragg_b[best] = mxv;
            avail_b = false;
            atomicAdd(&s_load[best], 1.0f);
        }
        __syncthreads();
        s_avail[b] = avail_b ? 1 : 0;
        __syncthreads();
    }

    if (avail_b) {
        int t = 0; float mx = init[0];
#pragma unroll
        for (int e = 1; e < 16; ++e)
            if (init[e] > mx) { mx = init[e]; t = e; }
        Dmask |= (1 << t);
        ragg_b[t] = init[t];
    }

    int cnt = __popc(Dmask);
    float wv = 1.f / fmaxf((float)cnt, 1.f);
    float* o_final = out;
    float* o_ragg  = out + 1024 * 10;
    float* o_D     = out + 1024 * 10 + 1024 * 16;
#pragma unroll
    for (int c = 0; c < 10; ++c) {
        float s = 0.f;
#pragma unroll
        for (int e = 0; e < 16; ++e)
            if (Dmask & (1 << e)) s += wv * le[e * 10 + c];
        o_final[b * 10 + c] = s;
    }
#pragma unroll
    for (int e = 0; e < 16; ++e) {
        o_ragg[b * 16 + e] = ragg_b[e];
        o_D[b * 16 + e]    = (Dmask >> e) & 1 ? 1.f : 0.f;
    }
}

// ---------------------------------------------------------------------------
// Orchestration
// ---------------------------------------------------------------------------
extern "C" void kernel_launch(void* const* d_in, const int* in_sizes, int n_in,
                              void* d_out, int out_size, void* d_ws, size_t ws_size,
                              hipStream_t stream)
{
    (void)in_sizes; (void)n_in; (void)out_size; (void)ws_size;

    const float* x    = (const float*)d_in[0];
    const float* c1w  = (const float*)d_in[1];
    const float* c1b  = (const float*)d_in[2];
    const float* bn1g = (const float*)d_in[3];
    const float* bn1b = (const float*)d_in[4];
    const float* bn1m = (const float*)d_in[5];
    const float* bn1v = (const float*)d_in[6];
    const float* c2w  = (const float*)d_in[7];
    const float* c2b  = (const float*)d_in[8];
    const float* bn2g = (const float*)d_in[9];
    const float* bn2b = (const float*)d_in[10];
    const float* bn2m = (const float*)d_in[11];
    const float* bn2v = (const float*)d_in[12];
    const float* c3w  = (const float*)d_in[13];
    const float* c3b  = (const float*)d_in[14];
    const float* bn3g = (const float*)d_in[15];
    const float* bn3b = (const float*)d_in[16];
    const float* bn3m = (const float*)d_in[17];
    const float* bn3v = (const float*)d_in[18];
    const float* c4w  = (const float*)d_in[19];
    const float* c4b  = (const float*)d_in[20];
    const float* bn4g = (const float*)d_in[21];
    const float* bn4b = (const float*)d_in[22];
    const float* bn4m = (const float*)d_in[23];
    const float* bn4v = (const float*)d_in[24];
    const float* g1w  = (const float*)d_in[25];
    const float* g1b  = (const float*)d_in[26];
    const float* g2w  = (const float*)d_in[27];
    const float* g2b  = (const float*)d_in[28];
    const float* clsw = (const float*)d_in[29];
    const float* clsb = (const float*)d_in[30];

    // workspace layout (floats)
    float* bufA     = (float*)d_ws;                  // 64M floats
    float* bufB     = bufA + (size_t)67108864;       // 64M floats
    float* feats    = bufB + (size_t)67108864;       // 1024*256
    float* gate_h   = feats + 262144;                // 1024*128
    float* gate_l   = gate_h + 131072;               // 1024*16
    float* logits_e = gate_l + 16384;                // 1024*160

    // bf16 fragment-ordered weight images (after logits_e, which is 163840 f32)
    unsigned short* w1i = (unsigned short*)(logits_e + 163840);
    unsigned short* w2i = w1i + 4096;      // conv1: NB=1, KSE=2
    unsigned short* w3i = w2i + 20480;     // conv2: NB=1, KSE=10
    unsigned short* w4i = w3i + 73728;     // conv3: NB=2, KSE=18
    unsigned short* g1i = w4i + 294912;    // conv4: NB=4, KSE=36
    unsigned short* g2i = g1i + 32768;     // g1:    NB=2, KSE=8
    unsigned short* cli = g2i + 8192;      // g2:    NB=1, KSE=4
                                           // cls:   NB=3, KSE=8 (49152)

    // build weight images (tiny)
    build_bimg<<<   4096 / 256, 256, 0, stream>>>(c1w,  w1i,  32,   27,  2, 1);
    build_bimg<<<  20480 / 256, 256, 0, stream>>>(c2w,  w2i,  64,  288, 10, 1);
    build_bimg<<<  73728 / 256, 256, 0, stream>>>(c3w,  w3i, 128,  576, 18, 1);
    build_bimg<<< 294912 / 256, 256, 0, stream>>>(c4w,  w4i, 256, 1152, 36, 1);
    build_bimg<<<  32768 / 256, 256, 0, stream>>>(g1w,  g1i, 128,  256,  8, 0);
    build_bimg<<<   8192 / 256, 256, 0, stream>>>(g2w,  g2i,  16,  128,  4, 0);
    build_bimg<<<  49152 / 256, 256, 0, stream>>>(clsw, cli, 160,  256,  8, 1);

    // trunk
    conv_bn_relu_wmma<32, 32, 3, 32><<<dim3(8192, 1), 256, 0, stream>>>(
        x, w1i, c1b, bn1g, bn1b, bn1m, bn1v, bufA);
    conv_bn_relu_wmma<32, 32, 32, 64><<<dim3(8192, 1), 256, 0, stream>>>(
        bufA, w2i, c2b, bn2g, bn2b, bn2m, bn2v, bufB);
    maxpool2<<<65536, 256, 0, stream>>>(bufB, bufA, 64, 32, 32, 1024 * 64 * 16 * 16);
    conv_bn_relu_wmma<16, 16, 64, 128><<<dim3(2048, 2), 256, 0, stream>>>(
        bufA, w3i, c3b, bn3g, bn3b, bn3m, bn3v, bufB);
    conv_bn_relu_wmma<16, 16, 128, 256><<<dim3(2048, 4), 256, 0, stream>>>(
        bufB, w4i, c4b, bn4g, bn4b, bn4m, bn4v, bufA);
    maxpool2<<<65536, 256, 0, stream>>>(bufA, bufB, 256, 16, 16, 1024 * 256 * 8 * 8);
    avgpool8<<<1024, 256, 0, stream>>>(bufB, feats);

    // gate MLP + per-expert classifier logits
    gemm_wmma<true ><<<dim3(8, 2), 256, 0, stream>>>(feats,  g1i, g1b,  gate_h,   1024, 128, 256, 8);
    gemm_wmma<false><<<dim3(8, 1), 256, 0, stream>>>(gate_h, g2i, g2b,  gate_l,   1024,  16, 128, 4);
    gemm_wmma<false><<<dim3(8, 3), 256, 0, stream>>>(feats,  cli, clsb, logits_e, 1024, 160, 256, 8);

    // routing + final mixture (single workgroup, ~68KB dynamic LDS)
    routing_kernel<<<1, 1024, 69888, stream>>>(gate_l, logits_e, (float*)d_out);
}
